// LSTMRecurrentPolicy_78692390797526
// MI455X (gfx1250) — matrix operand order repaired and checked by reading, not compiled, not run on hardware
//
#include <hip/hip_runtime.h>
#include <hip/hip_bf16.h>

// ---- CDNA5 WMMA vector types --------------------------------------------
typedef __attribute__((ext_vector_type(16))) __bf16 v16bf;
typedef __attribute__((ext_vector_type(8)))  __bf16 v8bf;
typedef __attribute__((ext_vector_type(8)))  float  v8f;

#define T_STEPS 512
#define BATCH   1024
#define CIN     128
#define HID     256
#define KTOT    (CIN + HID)      // 384 : unified K for [x | h] @ [W_ih | W_hh]^T
#define KSTEPS  (KTOT / 32)      // 12
#define ROWS    16               // batch rows per WG -> 64 WGs
#define NWAVES  16               // each wave owns 16 hidden units (i,f,g,o tiles)
#define THREADS (NWAVES * 32)    // 512 threads, wave32
#define APAD    8                // LDS row pad: stride 392 elems = 784B (bank rotate)
#define ASTRIDE (KTOT + APAD)

__device__ __forceinline__ float fast_sigmoid(float x) {
    return 1.0f / (1.0f + __expf(-x));
}
__device__ __forceinline__ float fast_tanh(float x) {
    x = fminf(fmaxf(x, -15.0f), 15.0f);
    float e = __expf(2.0f * x);
    return (e - 1.0f) / (e + 1.0f);
}

// ---- prep: swizzle [W_ih | W_hh] into B-fragment order, fuse biases ------
// Wfrag layout: [n_tile(64)][kk(12)][lane(32)][e(16)] bf16  (each kk block = 1KB,
// so one wave's per-k-step B load is a single contiguous, fully coalesced 1KB).
// Fragment mapping (16-bit B 32x16, contiguous k, same order as the A loads): lane%16 = N (W row), lane/16 selects
// K half: k = kk*32 + (lane/16)*16 + e.
__global__ void lstm_pack_kernel(const float* __restrict__ Wih,   // [4H, C]
                                 const float* __restrict__ Whh,   // [4H, H]
                                 const float* __restrict__ bih,   // [4H]
                                 const float* __restrict__ bhh,   // [4H]
                                 __bf16* __restrict__ Wfrag,      // [4H*KTOT]
                                 float* __restrict__ bias)        // [4H]
{
    const int stride = gridDim.x * blockDim.x;
    const int tid = blockIdx.x * blockDim.x + threadIdx.x;
    const int total = 4 * HID * KTOT;     // 393216
    for (int i = tid; i < total; i += stride) {
        int e    = i & 15;
        int ln   = (i >> 4) & 31;
        int rest = i >> 9;
        int kk   = rest % KSTEPS;
        int nt   = rest / KSTEPS;                  // 0..63
        int row  = nt * 16 + (ln & 15);            // W row (gate column)
        int k    = kk * 32 + (ln >> 4) * 16 + e;   // K index in [0, 384)
        float v = (k < CIN) ? Wih[row * CIN + k] : Whh[row * HID + (k - CIN)];
        Wfrag[i] = (__bf16)v;
    }
    for (int i = tid; i < 4 * HID; i += stride)
        bias[i] = bih[i] + bhh[i];
}

// ---- persistent recurrent kernel -----------------------------------------
__global__ __launch_bounds__(THREADS)
void lstm_recurrent_kernel(const float* __restrict__ x,          // [T,B,C]
                           const float* __restrict__ h0c0,       // [2,1,B,H]
                           const unsigned char* __restrict__ brk,// [T,B] (bool)
                           const __bf16* __restrict__ Wfrag,     // swizzled bf16
                           const float* __restrict__ bias,       // [4H]
                           float* __restrict__ out)              // [T,B,H]
{
    __shared__ __align__(32) __bf16 Ast[ROWS][ASTRIDE];  // [x | h] tile, ~12.5 KB

    const int tid  = threadIdx.x;
    const int lane = tid & 31;
    const int wave = tid >> 5;        // 0..15 : hidden units [16w, 16w+16)
    const int nloc = lane & 15;
    const int lhi  = lane >> 4;
    const int rowbase = blockIdx.x * ROWS;

    // Per-gate bias (column = g*H + 16*wave + nloc), broadcast over M
    float bgate[4];
#pragma unroll
    for (int g = 0; g < 4; ++g)
        bgate[g] = bias[g * HID + wave * 16 + nloc];

    // t-invariant B-fragment base pointers (one per gate tile of this wave)
    const __bf16* bptr[4];
#pragma unroll
    for (int g = 0; g < 4; ++g)
        bptr[g] = Wfrag + ((size_t)(g * 16 + wave) * KSTEPS) * 512 + lane * 16;

    // A-fragment base in LDS (lane -> row nloc, K-half by lhi)
    const __bf16* ap = &Ast[nloc][lhi * 16];

    // Cell state, D-fragment layout (lane: col nloc, rows lhi*8 + r)
    v8f cst;
#pragma unroll
    for (int r = 0; r < 8; ++r) {
        int grow = rowbase + lhi * 8 + r;
        cst[r] = h0c0[(size_t)BATCH * HID + (size_t)grow * HID + wave * 16 + nloc];
    }

    // Stage initial h into LDS columns [CIN, KTOT)
    for (int i = tid; i < ROWS * HID; i += THREADS) {
        int r = i / HID, c = i % HID;
        Ast[r][CIN + c] = (__bf16)h0c0[(size_t)(rowbase + r) * HID + c];
    }

#pragma unroll 1
    for (int t = 0; t < T_STEPS; ++t) {
        // Stage x[t] tile (f32 -> bf16) into LDS columns [0, CIN)
        const float* xt = x + (size_t)t * BATCH * CIN + (size_t)rowbase * CIN;
        for (int i = tid; i < ROWS * CIN; i += THREADS) {
            int r = i / CIN, c = i % CIN;
            Ast[r][c] = (__bf16)xt[r * CIN + c];
        }

        // Preload B for kk=0 while waiting on the barrier
        v16bf b0[4], b1[4];
#pragma unroll
        for (int g = 0; g < 4; ++g)
            b0[g] = *(const v16bf*)(bptr[g]);

        __syncthreads();   // A tile (x + h) fully staged

        v8f acc[4] = {};   // 4 gate tiles (i,f,g,o), f32 accumulators
        union AF { v16bf v; v8bf h[2]; } af;

        // Software-pipelined K loop: ping-pong B buffers, bounded live range
#pragma unroll 1
        for (int kk = 0; kk < KSTEPS; kk += 2) {
#pragma unroll
            for (int g = 0; g < 4; ++g)                     // prefetch kk+1
                b1[g] = *(const v16bf*)(bptr[g] + (kk + 1) * 512);
            af.h[0] = *(const v8bf*)(ap + kk * 32);
            af.h[1] = *(const v8bf*)(ap + kk * 32 + 8);
#pragma unroll
            for (int g = 0; g < 4; ++g)
                acc[g] = __builtin_amdgcn_wmma_f32_16x16x32_bf16(
                    false, af.v, false, b0[g], (short)0, acc[g], false, false);

            if (kk + 2 < KSTEPS) {
#pragma unroll
                for (int g = 0; g < 4; ++g)                 // prefetch kk+2
                    b0[g] = *(const v16bf*)(bptr[g] + (kk + 2) * 512);
            }
            af.h[0] = *(const v8bf*)(ap + (kk + 1) * 32);
            af.h[1] = *(const v8bf*)(ap + (kk + 1) * 32 + 8);
#pragma unroll
            for (int g = 0; g < 4; ++g)
                acc[g] = __builtin_amdgcn_wmma_f32_16x16x32_bf16(
                    false, af.v, false, b1[g], (short)0, acc[g], false, false);
        }
        __syncthreads();   // all waves done reading Ast; safe to overwrite h

        // Elementwise LSTM cell (wave-local i,f,g,o for our 16 units)
#pragma unroll
        for (int r = 0; r < 8; ++r) {
            const int mrow = lhi * 8 + r;
            const int grow = rowbase + mrow;
            float iv = acc[0][r] + bgate[0];
            float fv = acc[1][r] + bgate[1];
            float gv = acc[2][r] + bgate[2];
            float ov = acc[3][r] + bgate[3];
            float cn = fast_sigmoid(fv) * cst[r]
                     + fast_sigmoid(iv) * fast_tanh(gv);
            float hn = fast_sigmoid(ov) * fast_tanh(cn);
            out[(size_t)t * BATCH * HID + (size_t)grow * HID + wave * 16 + nloc] = hn;
            // reset AFTER emitting output
            float keep = brk[(size_t)t * BATCH + grow] ? 0.0f : 1.0f;
            cst[r] = cn * keep;
            Ast[mrow][CIN + wave * 16 + nloc] = (__bf16)(hn * keep);
        }

        // Warm caches for next timestep's x tile (512 thr * 4 = 2048 = tile)
        if (t + 1 < T_STEPS)
            __builtin_prefetch(x + (size_t)(t + 1) * BATCH * CIN
                                 + (size_t)rowbase * CIN + tid * 4, 0, 3);
        // next iteration's barrier orders the h-writes before the K-loop reads
    }
}

extern "C" void kernel_launch(void* const* d_in, const int* in_sizes, int n_in,
                              void* d_out, int out_size, void* d_ws, size_t ws_size,
                              hipStream_t stream) {
    const float* x    = (const float*)d_in[0];            // [T,B,C]
    const float* h0c0 = (const float*)d_in[1];            // [2,1,B,H]
    const float* Wih  = (const float*)d_in[2];            // [4H,C]
    const float* Whh  = (const float*)d_in[3];            // [4H,H]
    const float* bih  = (const float*)d_in[4];            // [4H]
    const float* bhh  = (const float*)d_in[5];            // [4H]
    const unsigned char* brk = (const unsigned char*)d_in[6]; // [T,B] bool
    float* out = (float*)d_out;

    __bf16* Wfrag = (__bf16*)d_ws;                                    // 768 KB
    float*  bias  = (float*)((char*)d_ws + (size_t)4 * HID * KTOT * sizeof(__bf16));

    lstm_pack_kernel<<<384, 256, 0, stream>>>(Wih, Whh, bih, bhh, Wfrag, bias);
    lstm_recurrent_kernel<<<BATCH / ROWS, THREADS, 0, stream>>>(
        x, h0c0, brk, Wfrag, bias, out);
    (void)in_sizes; (void)n_in; (void)out_size; (void)ws_size;
}
